// SGC_49778670960917
// MI455X (gfx1250) — matrix operand order repaired and checked by reading, not compiled, not run on hardware
//
#include <hip/hip_runtime.h>

// ---------------- problem constants (match reference) ----------------
#define NN      100000      // nodes
#define NE      1600000     // edges
#define IF      128         // in feats
#define NH      512         // hidden
#define NC      64          // classes
#define BN_EPS  1e-5f

// ---------------- workspace layout (bytes) ----------------
#define SZ_FT   ((size_t)NN * IF * 4)        // 51,200,000  (f32 node feats)
#define SZ_H    ((size_t)NN * NH * 2)        // 102,400,000 (bf16 hidden)
#define OFF_FTA ((size_t)0)
#define OFF_FTB (OFF_FTA + SZ_FT)            // also reused as bf16 feature buffer
#define OFF_H   (OFF_FTB + SZ_FT)
#define OFF_W1  (OFF_H + SZ_H)               // bf16 512x128
#define OFF_W2  (OFF_W1 + (size_t)NH * IF * 2)
#define OFF_SUM (OFF_W2 + (size_t)NC * NH * 2)
#define OFF_SQ  (OFF_SUM + (size_t)NH * 4)
#define OFF_SC  (OFF_SQ  + (size_t)NH * 4)
#define OFF_SH  (OFF_SC  + (size_t)NH * 4)

typedef __attribute__((ext_vector_type(16))) __bf16        v16bf;
typedef __attribute__((ext_vector_type(8)))  float         v8f;
typedef __attribute__((ext_vector_type(4)))  unsigned int  v4u;
typedef __attribute__((ext_vector_type(8)))  int           v8i;
typedef __attribute__((ext_vector_type(4)))  int           v4i;

union BF16x16 { v16bf v; unsigned short u[16]; };

__device__ __forceinline__ unsigned short f32_to_bf16(float f) {
  unsigned int x = __float_as_uint(f);
  unsigned int r = x + 0x7fffu + ((x >> 16) & 1u);   // round-to-nearest-even
  return (unsigned short)(r >> 16);
}
__device__ __forceinline__ float bf16_to_f32(unsigned short u) {
  return __uint_as_float(((unsigned int)u) << 16);
}

// ---------------- Tensor Data Mover: 2D bf16 tile -> LDS ----------------
// D# per CDNA5 ISA. data_size=1 (2 bytes).
// dim0 = tensor length along contiguous axis (elements), dim1 = #rows,
// tile0/tile1 = tile extents, stride0 = row stride (elements).
// This toolchain exposes the 6-arg builtin: (v4u, v8i, v4i, v4i, v8i, cpol).
__device__ __forceinline__ void tdm_load_2d_bf16(const void* gptr, unsigned ldsAddr,
                                                 unsigned dim0, unsigned dim1,
                                                 unsigned tile0, unsigned tile1,
                                                 unsigned stride0) {
  unsigned long long ga = (unsigned long long)(uintptr_t)gptr;
  v4u g0 = { 1u,                                     // count=1, is_restore=0
             ldsAddr,                                // lds_addr (bytes)
             (unsigned)ga,                           // global_addr[31:0]
             (unsigned)((ga >> 32) & 0x01FFFFFFull)  // global_addr[56:32]
               | (2u << 30) };                       // type=2 ("image")
  v8i g1 = { (int)(1u << 16),                        // wg_mask=0, data_size=1 (2B)
             (int)(dim0 << 16),                      // tensor_dim0[15:0] @ bits 63:48
             (int)((dim0 >> 16) | (dim1 << 16)),     // dim0 hi | dim1 lo
             (int)((dim1 >> 16) | (tile0 << 16)),    // dim1 hi | tile_dim0
             (int)(tile1 & 0xFFFFu),                 // tile_dim1 | tile_dim2=0
             (int)stride0,                           // tensor_dim0_stride[31:0]
             0, 0 };                                 // stride0 hi | stride1 = 0
  v4i z4 = { 0, 0, 0, 0 };
  v8i z8 = { 0, 0, 0, 0, 0, 0, 0, 0 };
  __builtin_amdgcn_tensor_load_to_lds(g0, g1, z4, z4, z8, 0);
}

// ---------------- utility kernels ----------------
__global__ void sgc_zero_f32(float* __restrict__ p, long n) {
  long i = (long)blockIdx.x * blockDim.x + threadIdx.x;
  long stride = (long)gridDim.x * blockDim.x;
  for (; i < n; i += stride) p[i] = 0.0f;
}

__global__ void sgc_cvt_bf16(const float* __restrict__ in, unsigned short* __restrict__ out, int n) {
  int i = blockIdx.x * blockDim.x + threadIdx.x;
  if (i < n) out[i] = f32_to_bf16(in[i]);
}

// ---------------- edge scatter: ft_out[dst] += ft_in[src] * w ----------------
__global__ __launch_bounds__(256)
void sgc_scatter(const float* __restrict__ fin, const int* __restrict__ src,
                 const int* __restrict__ dst, const float* __restrict__ nrm,
                 float* __restrict__ fout) {
  long gid = (long)blockIdx.x * blockDim.x + threadIdx.x;
  int e = (int)(gid >> 5);
  if (e >= NE) return;
  int c = ((int)gid & 31) << 2;                // feature offset 0..124
  float w = nrm[e];
  const float* sp = fin + (size_t)src[e] * IF + c;
  float*       dp = fout + (size_t)dst[e] * IF + c;
  float4 v = *(const float4*)sp;
  atomicAdd(dp + 0, v.x * w);
  atomicAdd(dp + 1, v.y * w);
  atomicAdd(dp + 2, v.z * w);
  atomicAdd(dp + 3, v.w * w);
}

// ---------------- GEMM1: h[N,512] = ft[N,128] @ W1^T + b1  (bf16 WMMA, f32 acc) ----------------
// block = 256 threads = 8 waves; wave -> 16 rows x 64 cols (4 N-tiles), block -> 128 rows.
// W1 block tile (64 x 128 bf16, 16KB) staged in LDS by the Tensor Data Mover.
__global__ __launch_bounds__(256)
void sgc_gemm1(const unsigned short* __restrict__ ftbf, const unsigned short* __restrict__ w1,
               const float* __restrict__ b1, unsigned short* __restrict__ h) {
  __shared__ unsigned short sB[64 * IF];       // 16 KB
  const int lane = threadIdx.x & 31;
  const int wave = threadIdx.x >> 5;
  const int rowBase = blockIdx.x * 128 + wave * 16;
  const int nBase   = blockIdx.y * 64;

  if (wave == 0) {
    tdm_load_2d_bf16(w1 + (size_t)nBase * IF,
                     (unsigned)(uintptr_t)(void*)sB,
                     /*dim0*/ IF, /*dim1*/ NH - nBase,
                     /*tile0*/ IF, /*tile1*/ 64, /*stride0*/ IF);
    __builtin_amdgcn_s_wait_tensorcnt(0);
  }
  __syncthreads();

  const int m   = lane & 15;
  const int hi  = lane >> 4;                   // lane half
  const int row = rowBase + m;
  const int rowc = row < NN ? row : (NN - 1);
  const unsigned short* ap = ftbf + (size_t)rowc * IF;

  v8f acc[4] = {};
  #pragma unroll
  for (int k0 = 0; k0 < IF; k0 += 32) {
    // A fragment: 16x32 bf16. lanes 0-15: K k0+0..15 ; lanes 16-31: K k0+16..31
    BF16x16 A;
    const int kA = k0 + 16 * hi;
    const int kB = k0 + 16 * hi + 8;
    __builtin_prefetch(ap + kA + 32, 0, 3);
    #pragma unroll
    for (int i = 0; i < 8; ++i) {
      A.u[i]     = ap[kA + i];
      A.u[8 + i] = ap[kB + i];
    }
    const int col  = lane & 15;
    const int koff = k0 + hi * 16;
    #pragma unroll
    for (int t = 0; t < 4; ++t) {
      const unsigned short* bp = sB + (size_t)(16 * t + col) * IF + koff;
      BF16x16 B;
      #pragma unroll
      for (int i = 0; i < 16; ++i) B.u[i] = bp[i];
      acc[t] = __builtin_amdgcn_wmma_f32_16x16x32_bf16(
          false, A.v, false, B.v, (short)0, acc[t], false, false);
    }
  }
  // store: lane<16 -> M=r ; lane>=16 -> M=8+r ; N = lane&15
  const int n15 = lane & 15;
  const int mrow = rowBase + 8 * hi;
  #pragma unroll
  for (int t = 0; t < 4; ++t) {
    const int n = nBase + 16 * t + n15;
    const float bias = b1[n];
    #pragma unroll
    for (int r = 0; r < 8; ++r) {
      const int rr = mrow + r;
      if (rr < NN) h[(size_t)rr * NH + n] = f32_to_bf16(acc[t][r] + bias);
    }
  }
}

// ---------------- BN stats: per-channel sum / sumsq over N rows ----------------
#define STAT_ROWS 500
__global__ __launch_bounds__(512)
void sgc_stats(const unsigned short* __restrict__ h, float* __restrict__ sum,
               float* __restrict__ sq) {
  const int t = threadIdx.x;               // channel 0..511
  const int r0 = blockIdx.x * STAT_ROWS;
  int r1 = r0 + STAT_ROWS; if (r1 > NN) r1 = NN;
  float s = 0.0f, q = 0.0f;
  for (int r = r0; r < r1; ++r) {
    float f = bf16_to_f32(h[(size_t)r * NH + t]);
    s += f; q += f * f;
  }
  atomicAdd(&sum[t], s);
  atomicAdd(&sq[t], q);
}

__global__ __launch_bounds__(512)
void sgc_bnfin(const float* __restrict__ sum, const float* __restrict__ sq,
               const float* __restrict__ gamma, const float* __restrict__ beta,
               float* __restrict__ scale, float* __restrict__ shift) {
  const int t = threadIdx.x;
  const float inv = 1.0f / (float)NN;
  float mean = sum[t] * inv;
  float var  = sq[t] * inv - mean * mean;
  float sc   = gamma[t] * rsqrtf(var + BN_EPS);
  scale[t] = sc;
  shift[t] = beta[t] - mean * sc;
}

// ---------------- GEMM2: out[N,64] = relu(h*scale+shift) @ W2^T + b2 ----------------
// Full W2 (64 x 512 bf16, 64KB) staged in LDS by the Tensor Data Mover.
__global__ __launch_bounds__(256)
void sgc_gemm2(const unsigned short* __restrict__ h, const unsigned short* __restrict__ w2,
               const float* __restrict__ b2, const float* __restrict__ scale,
               const float* __restrict__ shift, float* __restrict__ out) {
  __shared__ unsigned short sB[NC * NH];       // 64 KB
  const int lane = threadIdx.x & 31;
  const int wave = threadIdx.x >> 5;
  const int rowBase = blockIdx.x * 128 + wave * 16;

  if (wave == 0) {
    tdm_load_2d_bf16(w2, (unsigned)(uintptr_t)(void*)sB,
                     /*dim0*/ NH, /*dim1*/ NC,
                     /*tile0*/ NH, /*tile1*/ NC, /*stride0*/ NH);
    __builtin_amdgcn_s_wait_tensorcnt(0);
  }
  __syncthreads();

  const int m  = lane & 15;
  const int hi = lane >> 4;
  const int row = rowBase + m;
  const int rowc = row < NN ? row : (NN - 1);
  const unsigned short* ap = h + (size_t)rowc * NH;

  v8f acc[4] = {};
  #pragma unroll 4
  for (int k0 = 0; k0 < NH; k0 += 32) {
    BF16x16 A;
    const int kA = k0 + 16 * hi;
    const int kB = k0 + 16 * hi + 8;
    __builtin_prefetch(ap + kA + 32, 0, 3);
    #pragma unroll
    for (int i = 0; i < 8; ++i) {
      float fa = bf16_to_f32(ap[kA + i]);
      fa = fmaxf(fmaf(fa, scale[kA + i], shift[kA + i]), 0.0f);
      A.u[i] = f32_to_bf16(fa);
      float fb = bf16_to_f32(ap[kB + i]);
      fb = fmaxf(fmaf(fb, scale[kB + i], shift[kB + i]), 0.0f);
      A.u[8 + i] = f32_to_bf16(fb);
    }
    const int col  = lane & 15;
    const int koff = k0 + hi * 16;
    #pragma unroll
    for (int t = 0; t < 4; ++t) {
      const unsigned short* bp = sB + (size_t)(16 * t + col) * NH + koff;
      BF16x16 B;
      #pragma unroll
      for (int i = 0; i < 16; ++i) B.u[i] = bp[i];
      acc[t] = __builtin_amdgcn_wmma_f32_16x16x32_bf16(
          false, A.v, false, B.v, (short)0, acc[t], false, false);
    }
  }
  const int n15 = lane & 15;
  const int mrow = rowBase + 8 * hi;
  #pragma unroll
  for (int t = 0; t < 4; ++t) {
    const int n = 16 * t + n15;
    const float bias = b2[n];
    #pragma unroll
    for (int r = 0; r < 8; ++r) {
      const int rr = mrow + r;
      if (rr < NN) out[(size_t)rr * NC + n] = acc[t][r] + bias;
    }
  }
}

// ---------------- launcher ----------------
extern "C" void kernel_launch(void* const* d_in, const int* in_sizes, int n_in,
                              void* d_out, int out_size, void* d_ws, size_t ws_size,
                              hipStream_t stream) {
  const float* feat  = (const float*)d_in[0];
  const int*   src   = (const int*)  d_in[1];
  const int*   dst   = (const int*)  d_in[2];
  const float* nrm   = (const float*)d_in[3];
  const float* fc1w  = (const float*)d_in[4];
  const float* fc1b  = (const float*)d_in[5];
  const float* gamma = (const float*)d_in[6];
  const float* beta  = (const float*)d_in[7];
  const float* fc2w  = (const float*)d_in[8];
  const float* fc2b  = (const float*)d_in[9];
  float* out = (float*)d_out;

  char* ws = (char*)d_ws;
  float*          ftA  = (float*)(ws + OFF_FTA);
  float*          ftB  = (float*)(ws + OFF_FTB);
  unsigned short* ftbf = (unsigned short*)(ws + OFF_FTB);  // reuse after hop 3
  unsigned short* hbuf = (unsigned short*)(ws + OFF_H);
  unsigned short* w1b  = (unsigned short*)(ws + OFF_W1);
  unsigned short* w2b  = (unsigned short*)(ws + OFF_W2);
  float*          sum  = (float*)(ws + OFF_SUM);
  float*          sq   = (float*)(ws + OFF_SQ);
  float*          sc   = (float*)(ws + OFF_SC);
  float*          sh   = (float*)(ws + OFF_SH);

  const long nft = (long)NN * IF;
  const int scatterBlocks = (int)(((long)NE * 32 + 255) / 256);

  // hop 1: feat -> ftA
  sgc_zero_f32<<<2048, 256, 0, stream>>>(ftA, nft);
  sgc_scatter<<<scatterBlocks, 256, 0, stream>>>(feat, src, dst, nrm, ftA);
  // hop 2: ftA -> ftB
  sgc_zero_f32<<<2048, 256, 0, stream>>>(ftB, nft);
  sgc_scatter<<<scatterBlocks, 256, 0, stream>>>(ftA, src, dst, nrm, ftB);
  // hop 3: ftB -> ftA
  sgc_zero_f32<<<2048, 256, 0, stream>>>(ftA, nft);
  sgc_scatter<<<scatterBlocks, 256, 0, stream>>>(ftB, src, dst, nrm, ftA);

  // convert propagated features + weights to bf16 (ftB region reused for bf16 feats)
  sgc_cvt_bf16<<<(int)((nft + 255) / 256), 256, 0, stream>>>(ftA, ftbf, (int)nft);
  sgc_cvt_bf16<<<(NH * IF + 255) / 256, 256, 0, stream>>>(fc1w, w1b, NH * IF);
  sgc_cvt_bf16<<<(NC * NH + 255) / 256, 256, 0, stream>>>(fc2w, w2b, NC * NH);

  // zero stats (sum + sq contiguous)
  sgc_zero_f32<<<4, 256, 0, stream>>>(sum, NH * 2L);

  // GEMM1 + bias -> h (bf16)
  dim3 g1((NN + 127) / 128, NH / 64);
  sgc_gemm1<<<g1, 256, 0, stream>>>(ftbf, w1b, fc1b, hbuf);

  // BN stats + finalize
  sgc_stats<<<(NN + STAT_ROWS - 1) / STAT_ROWS, 512, 0, stream>>>(hbuf, sum, sq);
  sgc_bnfin<<<1, 512, 0, stream>>>(sum, sq, gamma, beta, sc, sh);

  // GEMM2 with fused BN + ReLU -> out (f32)
  sgc_gemm2<<<(NN + 127) / 128, 256, 0, stream>>>(hbuf, w2b, fc2b, sc, sh, out);
}
